// Head_9826885173975
// MI455X (gfx1250) — matrix operand order repaired and checked by reading, not compiled, not run on hardware
//
#include <hip/hip_runtime.h>

#define BATCH 4
#define SEQ   4096
#define INDIM 1024
#define LAT   128

typedef __attribute__((ext_vector_type(16))) __bf16 v16bf;
typedef __attribute__((ext_vector_type(8)))  float  v8f;

struct Bytes32 { uint4 lo; uint4 hi; };

// Build a v16bf A/B fragment from two 16-byte chunks (elements 0..7 / 8..15).
__device__ __forceinline__ v16bf load_frag(const void* p0, const void* p1) {
  Bytes32 t;
  t.lo = *reinterpret_cast<const uint4*>(p0);
  t.hi = *reinterpret_cast<const uint4*>(p1);
  return __builtin_bit_cast(v16bf, t);
}

__device__ __forceinline__ v8f wmma_bf16(v16bf a, v16bf b, v8f c) {
  return __builtin_amdgcn_wmma_f32_16x16x32_bf16(false, a, false, b, (short)0, c,
                                                 false, false);
}

// XOR-16 lane swap (SWAPX16: and=0x1f, or=0, xor=0x10 -> 0x401F)
__device__ __forceinline__ float xor16f(float v) {
  return __int_as_float(__builtin_amdgcn_ds_swizzle(__float_as_int(v), 0x401F));
}
// Broadcast value from srcLane to all lanes (wave32 ds_bpermute, byte index)
__device__ __forceinline__ float bcastf(float v, int srcLane) {
  return __int_as_float(__builtin_amdgcn_ds_bpermute(srcLane << 2, __float_as_int(v)));
}

// ---------------------------------------------------------------------------
// Projection: out[M,128] = in[M,Kdim] @ W[128,Kdim]^T + bias, out in bf16.
// Block = 256 threads (8 waves); block tile = 128 rows x 128 cols.
// Wave w owns rows m0..m0+15 (A fragments), all 128 output cols (8 WMMA accums).
// W k-chunk staged in LDS as bf16, laid out for direct B-fragment reads.
// TRANS_OUT stores out transposed per batch: VpT[b][o][n] (for attention B frags).
// ---------------------------------------------------------------------------
template <bool IN_BF16, bool TRANS_OUT>
__global__ __launch_bounds__(256) void proj_kernel(
    const void* __restrict__ in_, const float* __restrict__ W,
    const float* __restrict__ bias, __bf16* __restrict__ out, int Kdim) {
  __shared__ __align__(16) unsigned char wlds[LAT * 80];  // [o][32k] bf16, 80B stride

  const int tid  = threadIdx.x;
  const int lane = tid & 31;
  const int wave = tid >> 5;
  const int l15  = lane & 15;
  const int hh   = lane >> 4;  // 0: lanes 0-15, 1: lanes 16-31
  const int m0   = blockIdx.x * 128 + wave * 16;

  v8f acc[8];
#pragma unroll
  for (int i = 0; i < 8; ++i)
#pragma unroll
    for (int e = 0; e < 8; ++e) acc[i][e] = 0.f;

  const int nChunks = Kdim >> 5;
  for (int kc = 0; kc < nChunks; ++kc) {
    const int k0 = kc << 5;
    {  // stage W[*, k0..k0+31] as bf16 into LDS
      const int o  = tid >> 1;
      const int ks = (tid & 1) << 4;  // element offset 0 or 16
      const float* wp = W + (size_t)o * Kdim + k0 + ks;
      float4 f0 = *(const float4*)(wp + 0);
      float4 f1 = *(const float4*)(wp + 4);
      float4 f2 = *(const float4*)(wp + 8);
      float4 f3 = *(const float4*)(wp + 12);
      union { __bf16 h[16]; uint4 u[2]; } pk;
      float fv[16] = {f0.x, f0.y, f0.z, f0.w, f1.x, f1.y, f1.z, f1.w,
                      f2.x, f2.y, f2.z, f2.w, f3.x, f3.y, f3.z, f3.w};
#pragma unroll
      for (int i = 0; i < 16; ++i) pk.h[i] = (__bf16)fv[i];
      uint4* dst = (uint4*)(wlds + o * 80 + (ks << 1));
      dst[0] = pk.u[0];
      dst[1] = pk.u[1];
    }
    __syncthreads();

    // A fragment (16x32, 16-bit layout): lanes<16 k:{0-7,16-23}, lanes>=16 k:{8-15,24-31}
    v16bf afrag;
    if (IN_BF16) {
      const __bf16* ip =
          (const __bf16*)in_ + (size_t)(m0 + l15) * Kdim + k0 + (hh ? 8 : 0);
      afrag = load_frag(ip, ip + 16);
    } else {
      const float* ip =
          (const float*)in_ + (size_t)(m0 + l15) * Kdim + k0 + (hh ? 8 : 0);
      float4 f0 = *(const float4*)(ip + 0);
      float4 f1 = *(const float4*)(ip + 4);
      float4 f2 = *(const float4*)(ip + 16);
      float4 f3 = *(const float4*)(ip + 20);
      union { __bf16 h[16]; v16bf v; } cv;
      float fv[16] = {f0.x, f0.y, f0.z, f0.w, f1.x, f1.y, f1.z, f1.w,
                      f2.x, f2.y, f2.z, f2.w, f3.x, f3.y, f3.z, f3.w};
#pragma unroll
      for (int i = 0; i < 16; ++i) cv.h[i] = (__bf16)fv[i];
      afrag = cv.v;
    }

#pragma unroll
    for (int ot = 0; ot < 8; ++ot) {
      // B fragment (32x16): lane = out col, lanes<16 hold k {0-7,16-23}, lanes>=16 k {8-15,24-31}
      const unsigned char* bp = wlds + (ot * 16 + l15) * 80 + (hh ? 16 : 0);
      v16bf bfrag = load_frag(bp, bp + 32);
      acc[ot] = wmma_bf16(afrag, bfrag, acc[ot]);
    }
    __syncthreads();
  }

#pragma unroll
  for (int ot = 0; ot < 8; ++ot) {
    const int o    = ot * 16 + l15;
    const float bv = bias[o];
    if (!TRANS_OUT) {
#pragma unroll
      for (int j = 0; j < 8; ++j) {
        const int row = m0 + j + (hh ? 8 : 0);  // D: VGPR j -> row j (+8 hi half)
        out[(size_t)row * LAT + o] = (__bf16)(acc[ot][j] + bv);
      }
    } else {
      union { __bf16 h[8]; uint4 u; } pk;
#pragma unroll
      for (int j = 0; j < 8; ++j) pk.h[j] = (__bf16)(acc[ot][j] + bv);
      const int mrow0 = m0 + (hh ? 8 : 0);
      const int bidx  = mrow0 >> 12;        // / SEQ
      const int n     = mrow0 & (SEQ - 1);  // 8 consecutive n -> one b128 store
      *(uint4*)(out + ((size_t)bidx * LAT + o) * SEQ + n) = pk.u;
    }
  }
}

// ---------------------------------------------------------------------------
// Attention: Y[b,m,:] = sum_n softmax_n(Kp[n]·Qp[m]/64) * Vp[n,:]
// Block = 8 waves, covers 128 m-columns; wave w owns 16 m, full d=128.
// Score tiles computed as D[n,m] = Kp(A) x Qp^T(B): n on VGPR axis (softmax
// within lane), m on lane axis. Exp'd tiles feed the P x Vp WMMA A operand
// with no cross-lane movement. Kp / VpT chunks staged in LDS per block.
// ---------------------------------------------------------------------------
__global__ __launch_bounds__(256) void attn_kernel(const __bf16* __restrict__ Qp,
                                                   const __bf16* __restrict__ Kp,
                                                   const __bf16* __restrict__ VpT,
                                                   float* __restrict__ Y) {
  __shared__ __align__(16) unsigned char klds[32 * 272];   // Kp chunk, padded rows
  __shared__ __align__(16) unsigned char vlds[LAT * 80];   // VpT chunk, padded rows

  const int tid  = threadIdx.x;
  const int lane = tid & 31;
  const int wave = tid >> 5;
  const int l15  = lane & 15;
  const int hh   = lane >> 4;
  const int b    = blockIdx.y;
  const int m0   = blockIdx.x * 128 + wave * 16;

  // Qp^T B fragments (32d x 16m), one per d-chunk; contiguous row reads of Qp.
  v16bf qb[4];
  {
    const __bf16* q = Qp + ((size_t)b * SEQ + m0 + l15) * LAT + (hh ? 8 : 0);
#pragma unroll
    for (int dc = 0; dc < 4; ++dc) qb[dc] = load_frag(q + dc * 32, q + dc * 32 + 16);
  }

  v8f yacc[8];
#pragma unroll
  for (int i = 0; i < 8; ++i)
#pragma unroll
    for (int e = 0; e < 8; ++e) yacc[i][e] = 0.f;

  float run_max = -__builtin_inff();
  float run_sum = 0.f;
  const float scale = 1.0f / 64.0f;  // 1/sqrt(SEQ)

  const __bf16* kbase = Kp + (size_t)b * SEQ * LAT;
  const __bf16* vbase = VpT + (size_t)b * LAT * SEQ;

  const int kr = tid >> 3, koff = (tid & 7) << 5;  // Kp staging: 32 rows x 256B
  const int vr = tid >> 1, voff = (tid & 1) << 5;  // VpT staging: 128 rows x 64B

  for (int n0 = 0; n0 < SEQ; n0 += 32) {
    {  // stage Kp[n0..n0+31][0..127] (contiguous 8KB)
      const unsigned char* src =
          (const unsigned char*)(kbase + (size_t)(n0 + kr) * LAT) + koff;
      uint4 x0 = *(const uint4*)(src);
      uint4 x1 = *(const uint4*)(src + 16);
      uint4* dst = (uint4*)(klds + kr * 272 + koff);
      dst[0] = x0; dst[1] = x1;
    }
    {  // stage VpT[0..127][n0..n0+31]
      const unsigned char* src =
          (const unsigned char*)(vbase + (size_t)vr * SEQ + n0) + voff;
      uint4 x0 = *(const uint4*)(src);
      uint4 x1 = *(const uint4*)(src + 16);
      uint4* dst = (uint4*)(vlds + vr * 80 + voff);
      dst[0] = x0; dst[1] = x1;
    }
    if (n0 + 32 < SEQ) {  // prefetch next chunk into cache
      __builtin_prefetch(
          (const unsigned char*)(kbase + (size_t)(n0 + 32 + kr) * LAT) + koff, 0, 1);
      __builtin_prefetch(
          (const unsigned char*)(vbase + (size_t)vr * SEQ + n0 + 32) + voff, 0, 1);
    }
    __syncthreads();

    // S tiles: D[n,m], tile0 = n0..n0+15, tile1 = n0+16..n0+31
    v8f s0, s1;
#pragma unroll
    for (int e = 0; e < 8; ++e) { s0[e] = 0.f; s1[e] = 0.f; }
#pragma unroll
    for (int dc = 0; dc < 4; ++dc) {
      const unsigned char* a0 = klds + l15 * 272 + dc * 64 + (hh ? 16 : 0);
      const unsigned char* a1 = a0 + 16 * 272;
      v16bf ka0 = load_frag(a0, a0 + 32);
      v16bf ka1 = load_frag(a1, a1 + 32);
      s0 = wmma_bf16(ka0, qb[dc], s0);
      s1 = wmma_bf16(ka1, qb[dc], s1);
    }

    // Online softmax over n: per-lane stats (column m split across lane pair
    // l / l+16) merged with one XOR-16 swizzle.
    float p0[8], p1[8];
    float cmax = -__builtin_inff();
#pragma unroll
    for (int j = 0; j < 8; ++j) {
      p0[j] = s0[j] * scale;
      p1[j] = s1[j] * scale;
      cmax  = fmaxf(cmax, fmaxf(p0[j], p1[j]));
    }
    cmax = fmaxf(cmax, xor16f(cmax));
    const float nmax  = fmaxf(run_max, cmax);
    const float alpha = __expf(run_max - nmax);  // -inf first iter -> 0
    union { __bf16 h[16]; v16bf v; } pk;  // P fragment in A layout, directly
    float csum = 0.f;
#pragma unroll
    for (int j = 0; j < 8; ++j) {
      const float e0 = __expf(p0[j] - nmax);
      const float e1 = __expf(p1[j] - nmax);
      csum += e0 + e1;
      pk.h[j]     = (__bf16)e0;  // n-half from tile0
      pk.h[8 + j] = (__bf16)e1;  // n-half from tile1
    }
    csum += xor16f(csum);
    run_sum = run_sum * alpha + csum;
    run_max = nmax;

    // Rescale accumulators: Y row index = VGPR index, alpha lives per-lane.
#pragma unroll
    for (int j = 0; j < 8; ++j) {
      const float fl = bcastf(alpha, j);
      const float fh = bcastf(alpha, j + 8);
      const float f  = hh ? fh : fl;
#pragma unroll
      for (int dt = 0; dt < 8; ++dt) yacc[dt][j] *= f;
    }

    // Y += P(16m x 32n) x Vp(32n x 16d) for 8 d-tiles; B frags from VpT rows.
#pragma unroll
    for (int dt = 0; dt < 8; ++dt) {
      const unsigned char* bp = vlds + (dt * 16 + l15) * 80 + (hh ? 16 : 0);
      v16bf vb = load_frag(bp, bp + 32);
      yacc[dt] = wmma_bf16(pk.v, vb, yacc[dt]);
    }
    __syncthreads();
  }

  const float inv = 1.0f / run_sum;
#pragma unroll
  for (int j = 0; j < 8; ++j) {
    const float fl = bcastf(inv, j);
    const float fh = bcastf(inv, j + 8);
    const float f  = hh ? fh : fl;
    const int m    = m0 + j + (hh ? 8 : 0);
    float* yrow    = Y + ((size_t)b * SEQ + m) * LAT + l15;
#pragma unroll
    for (int dt = 0; dt < 8; ++dt) yrow[dt * 16] = yacc[dt][j] * f;
  }
}

extern "C" void kernel_launch(void* const* d_in, const int* in_sizes, int n_in,
                              void* d_out, int out_size, void* d_ws, size_t ws_size,
                              hipStream_t stream) {
  (void)in_sizes; (void)n_in; (void)out_size; (void)ws_size;
  const float* Q  = (const float*)d_in[0];
  // d_in[1] (K) intentionally unused: reference quirk ignores input K.
  const float* V  = (const float*)d_in[2];
  const float* Wk = (const float*)d_in[3];
  const float* bk = (const float*)d_in[4];
  const float* Wq = (const float*)d_in[5];
  const float* bq = (const float*)d_in[6];
  const float* Wv = (const float*)d_in[7];
  const float* bv = (const float*)d_in[8];
  float* Y = (float*)d_out;

  // Workspace: Kp | Qp | VpT, each B*N*128 bf16 = 4 MiB (12 MiB total).
  char* ws     = (char*)d_ws;
  __bf16* Kp   = (__bf16*)(ws);
  __bf16* Qp   = (__bf16*)(ws + (size_t)BATCH * SEQ * LAT * 2);
  __bf16* VpT  = (__bf16*)(ws + (size_t)BATCH * SEQ * LAT * 4);

  dim3 blk(256);
  proj_kernel<false, false><<<dim3(128), blk, 0, stream>>>(Q, Wk, bk, Kp, INDIM);
  proj_kernel<false, true ><<<dim3(128), blk, 0, stream>>>(V, Wv, bv, VpT, INDIM);
  proj_kernel<true,  false><<<dim3(128), blk, 0, stream>>>(Kp, Wq, bq, Qp, LAT);
  attn_kernel<<<dim3(SEQ / 128, BATCH), blk, 0, stream>>>(Qp, Kp, VpT, Y);
}
